// CausalGraphVAE_82480551952673
// MI455X (gfx1250) — compile-verified, matrix operand order checked by reading, not run on hardware
//
#include <hip/hip_runtime.h>
#include <hip/hip_bf16.h>
#include <math.h>

// Model dims (fixed by the reference)
#define Bn     512
#define Tn     32
#define Nn     512
#define Hn     64
#define Ln     128
#define NHEADS 4
#define Dn     64
#define L2n    64
#define G3H    192   // 3*H

typedef __attribute__((ext_vector_type(16))) _Float16 v16h;
typedef __attribute__((ext_vector_type(8)))  float    v8f;

__device__ __forceinline__ float sigm(float x) { return 1.0f / (1.0f + __expf(-x)); }

// Pack a 16-half WMMA fragment from fp32 memory (global or LDS).
// ISA 7.12.2 16-bit A/B 16x32 layout: lane&15 = row, lane>>4 = K-half;
// element j<8 -> k = 8*half + j ; j>=8 -> k = 16 + 8*half + (j-8).
// Caller passes p already offset by (row*stride + k0 + 8*half).
__device__ __forceinline__ v16h load_frag_f16(const float* __restrict__ p) {
  v16h f;
#pragma unroll
  for (int j = 0; j < 8; ++j) {
    f[j]     = (_Float16)p[j];
    f[j + 8] = (_Float16)p[16 + j];
  }
  return f;
}

// ---------------------------------------------------------------------------
// Generic WMMA GEMM: C[m,n] = sum_k A[m*lda + k] * W[n*K + k] + bias[n]
// (W stored [N,K] i.e. PyTorch weight layout, out = in @ W.T)
// One wave per 32x32 C tile: 2 A-frags x 2 B-frags -> 4 WMMAs per 32-K chunk
// (2x data reuse per fragment vs a 16x16 tile).
// Requirements: M,N multiples of 32; K multiple of 32.
// ---------------------------------------------------------------------------
__global__ __launch_bounds__(32) void gemm_bias_wmma(
    const float* __restrict__ A, int lda,
    const float* __restrict__ W,
    const float* __restrict__ bias,
    float* __restrict__ C,
    int M, int N, int K)
{
  const int lane = threadIdx.x;
  const int half = lane >> 4;
  const int mr   = lane & 15;
  const int n0 = blockIdx.x * 32;
  const int m0 = blockIdx.y * 32;

  const float* Arow0 = A + (size_t)(m0 + mr) * lda;
  const float* Arow1 = A + (size_t)(m0 + 16 + mr) * lda;
  const float* Wrow0 = W + (size_t)(n0 + mr) * K;
  const float* Wrow1 = W + (size_t)(n0 + 16 + mr) * K;

  v8f a00 = {}, a01 = {}, a10 = {}, a11 = {};
  for (int k0 = 0; k0 < K; k0 += 32) {
    const int ko = k0 + 8 * half;
    v16h af0 = load_frag_f16(Arow0 + ko);
    v16h af1 = load_frag_f16(Arow1 + ko);
    v16h bf0 = load_frag_f16(Wrow0 + ko);
    v16h bf1 = load_frag_f16(Wrow1 + ko);
    a00 = __builtin_amdgcn_wmma_f32_16x16x32_f16(false, af0, false, bf0, (short)0, a00, false, false);
    a01 = __builtin_amdgcn_wmma_f32_16x16x32_f16(false, af0, false, bf1, (short)0, a01, false, false);
    a10 = __builtin_amdgcn_wmma_f32_16x16x32_f16(false, af1, false, bf0, (short)0, a10, false, false);
    a11 = __builtin_amdgcn_wmma_f32_16x16x32_f16(false, af1, false, bf1, (short)0, a11, false, false);
  }

  const float bv0 = bias[n0 + mr];
  const float bv1 = bias[n0 + 16 + mr];
#pragma unroll
  for (int i = 0; i < 8; ++i) {
    const int mA = m0 + 8 * half + i;        // C layout: VGPR i -> M=i (+8 upper half)
    const int mB = mA + 16;
    C[(size_t)mA * N + n0 + mr]      = a00[i] + bv0;
    C[(size_t)mA * N + n0 + 16 + mr] = a01[i] + bv1;
    C[(size_t)mB * N + n0 + mr]      = a10[i] + bv0;
    C[(size_t)mB * N + n0 + 16 + mr] = a11[i] + bv1;
  }
}

// ---------------------------------------------------------------------------
// Fused full-sequence GRU. The recurrence is independent per batch row, so one
// block owns 16 batch rows and runs all Tn steps internally: h lives in LDS,
// GH = h @ Whh.T + bhh is computed per step by 4 waves (3 n-tiles each) with
// WMMA reading A fragments straight from LDS, then 128 threads apply the gates.
// Collapses 64 dependent kernel launches into 1.
//   encoder: gi_row_stride = Tn*G3H, gi_t_stride = G3H, store = nullptr
//   decoder: gi_row_stride = G3H,    gi_t_stride = 0,   store = dec_out
// ---------------------------------------------------------------------------
#define H_STRIDE  68    // 64 + 4 pad: lane r reads row r -> banks (4r + k) % 64
#define GH_STRIDE 196   // 192 + 4 pad

__global__ __launch_bounds__(128) void gru_seq_wmma(
    const float* __restrict__ gi, int gi_row_stride, int gi_t_stride,
    const float* __restrict__ Whh, const float* __restrict__ bhh,
    float* __restrict__ store,      // dec hidden states [b][t][j], or nullptr
    float* __restrict__ h_final)    // final hidden state [b][j], or nullptr
{
  __shared__ float hs[16 * H_STRIDE];
  __shared__ float ghs[16 * GH_STRIDE];

  const int tid  = threadIdx.x;
  const int wave = tid >> 5;
  const int lane = tid & 31;
  const int half = lane >> 4;
  const int mr   = lane & 15;
  const int m0   = blockIdx.x * 16;          // batch rows [m0, m0+16)

  for (int i = tid; i < 16 * H_STRIDE; i += 128) hs[i] = 0.0f;
  __syncthreads();

  for (int t = 0; t < Tn; ++t) {
    // --- GH strip: 16 x 192 via WMMA, A fragments from LDS-resident h ---
    const float* hrow = hs + mr * H_STRIDE;
#pragma unroll
    for (int s = 0; s < 3; ++s) {
      const int n0 = (wave * 3 + s) * 16;
      const float* Wrow = Whh + (size_t)(n0 + mr) * Hn;
      v8f acc = {};
#pragma unroll
      for (int k0 = 0; k0 < Hn; k0 += 32) {
        const int ko = k0 + 8 * half;
        v16h af = load_frag_f16(hrow + ko);
        v16h bf = load_frag_f16(Wrow + ko);
        acc = __builtin_amdgcn_wmma_f32_16x16x32_f16(false, af, false, bf, (short)0, acc, false, false);
      }
      const float bv = bhh[n0 + mr];
#pragma unroll
      for (int i = 0; i < 8; ++i)
        ghs[(8 * half + i) * GH_STRIDE + n0 + mr] = acc[i] + bv;
    }
    __syncthreads();

    // --- Gates: 16 rows x 64 cols, 8 elements per thread ---
    const float* git = gi + (size_t)t * gi_t_stride;
#pragma unroll
    for (int q = 0; q < 8; ++q) {
      const int lin = tid + q * 128;
      const int r = lin >> 6;
      const int j = lin & 63;
      const float* gib = git + (size_t)(m0 + r) * gi_row_stride;
      const float* ghr = ghs + r * GH_STRIDE;
      float rr = sigm(gib[j]          + ghr[j]);
      float zz = sigm(gib[Hn + j]     + ghr[Hn + j]);
      float nn = tanhf(gib[2 * Hn + j] + rr * ghr[2 * Hn + j]);
      float hv = hs[r * H_STRIDE + j];
      float hn = (1.0f - zz) * nn + zz * hv;
      hs[r * H_STRIDE + j] = hn;
      if (store) store[(size_t)(m0 + r) * (Tn * Hn) + (size_t)t * Hn + j] = hn;
    }
    __syncthreads();
  }

  if (h_final) {
#pragma unroll
    for (int q = 0; q < 8; ++q) {
      const int lin = tid + q * 128;
      const int r = lin >> 6;
      const int j = lin & 63;
      h_final[(size_t)(m0 + r) * Hn + j] = hs[r * H_STRIDE + j];
    }
  }
}

// ---------------------------------------------------------------------------
// Z = mu + eps1*exp(0.5*lv);  Z2 = mu + eps2*exp(0.5*lv)
__global__ void reparam(const float* __restrict__ mu, const float* __restrict__ lv,
                        const float* __restrict__ e1, const float* __restrict__ e2,
                        float* __restrict__ Z, float* __restrict__ Z2, int n)
{
  int i = blockIdx.x * blockDim.x + threadIdx.x;
  if (i >= n) return;
  float s = __expf(0.5f * lv[i]);
  float m = mu[i];
  Z[i]  = m + e1[i] * s;
  Z2[i] = m + e2[i] * s;
}

// adj = sigmoid(es + prior); clip [0,1]; zero diagonal; zeros -> 1e-8
__global__ void adj_kernel(const float* __restrict__ es, const float* __restrict__ prior,
                           float* __restrict__ out)
{
  int idx = blockIdx.x * blockDim.x + threadIdx.x;
  if (idx >= Nn * Nn) return;
  int i = idx / Nn, j = idx % Nn;
  float x = es[idx] + (prior ? prior[idx] : 0.0f);
  float a = sigm(x);
  a = fminf(fmaxf(a, 0.0f), 1.0f);
  if (i == j) a = 0.0f;
  if (a == 0.0f) a = 1e-8f;
  out[idx] = a;
}

// Flow + time-gate fusion: z2' = sigmoid(sraw)*z2 + traw + (1-zg)*n ; Zf=[z1,z2']
__global__ void flow_tg_kernel(const float* __restrict__ Z,
                               const float* __restrict__ sraw,
                               const float* __restrict__ traw,
                               const float* __restrict__ gitg,
                               const float* __restrict__ tg_bhh,
                               float* __restrict__ Zf)
{
  int idx = blockIdx.x * blockDim.x + threadIdx.x;
  if (idx >= Bn * L2n) return;
  int b = idx / L2n, j = idx % L2n;
  float z1 = Z[(size_t)b * Ln + j];
  float z2 = Z[(size_t)b * Ln + L2n + j];
  float s = sigm(sraw[(size_t)b * L2n + j]);
  float t = traw[(size_t)b * L2n + j];
  z2 = s * z2 + t;
  const float* gib = gitg + (size_t)b * G3H;
  float r  = sigm(gib[j] + tg_bhh[j]);
  float zg = sigm(gib[L2n + j] + tg_bhh[L2n + j]);
  float n  = tanhf(gib[2 * L2n + j] + r * tg_bhh[2 * L2n + j]);
  z2 += (1.0f - zg) * n;
  Zf[(size_t)b * Ln + j]       = z1;
  Zf[(size_t)b * Ln + L2n + j] = z2;
}

// e[h][i][j] = sum_d leaky_relu(xr[i,h,d] + xl[j,h,d], 0.2) * att[h,d]
__global__ void gat_e_kernel(const float* __restrict__ xl, const float* __restrict__ xr,
                             const float* __restrict__ att, float* __restrict__ e)
{
  int idx = blockIdx.x * blockDim.x + threadIdx.x;
  if (idx >= Bn * Bn * NHEADS) return;
  int j = idx % Bn;
  int i = (idx / Bn) % Bn;
  int h = idx / (Bn * Bn);
  const float* xri = xr + (size_t)i * (NHEADS * Dn) + h * Dn;
  const float* xlj = xl + (size_t)j * (NHEADS * Dn) + h * Dn;
  const float* ah  = att + h * Dn;
  float s = 0.0f;
#pragma unroll 4
  for (int d = 0; d < Dn; ++d) {
    float v = xri[d] + xlj[d];
    v = (v > 0.0f) ? v : 0.2f * v;
    s += v * ah[d];
  }
  e[idx] = s;
}

// Softmax over j per (h,i). One wave per contiguous row of 512, in place.
__global__ __launch_bounds__(32) void gat_softmax(float* __restrict__ e)
{
  float* base = e + (size_t)blockIdx.x * Bn;    // row = h*Bn + i
  int lane = threadIdx.x;
  float mx = -1e30f;
  for (int j = lane; j < Bn; j += 32) mx = fmaxf(mx, base[j]);
  for (int o = 16; o > 0; o >>= 1) mx = fmaxf(mx, __shfl_xor(mx, o, 32));
  float sum = 0.0f;
  for (int j = lane; j < Bn; j += 32) {
    float v = __expf(base[j] - mx);
    base[j] = v;
    sum += v;
  }
  for (int o = 16; o > 0; o >>= 1) sum += __shfl_xor(sum, o, 32);
  float inv = 1.0f / sum;
  for (int j = lane; j < Bn; j += 32) base[j] *= inv;
}

// X_emb[i, h*D+d] = sum_j alpha[h][i][j] * xl[j, h*D+d]  + gat_bias
__global__ void gat_aggregate(const float* __restrict__ alpha, const float* __restrict__ xl,
                              const float* __restrict__ gbias, float* __restrict__ xemb)
{
  int idx = blockIdx.x * blockDim.x + threadIdx.x;
  if (idx >= Bn * NHEADS * Dn) return;
  int i = idx / (NHEADS * Dn);
  int c = idx % (NHEADS * Dn);
  int h = c / Dn;
  const float* al = alpha + (size_t)h * Bn * Bn + (size_t)i * Bn;
  float s = 0.0f;
  for (int j = 0; j < Bn; ++j) s += al[j] * xl[(size_t)j * (NHEADS * Dn) + c];
  xemb[idx] = s + gbias[c];
}

// Final 2-wide projection: mean / std. One wave per batch row, lane reduction.
__global__ __launch_bounds__(32) void gl_kernel(const float* __restrict__ xemb,
                                                const float* __restrict__ Wm,
                                                const float* __restrict__ bm,
                                                float* __restrict__ mean_o,
                                                float* __restrict__ std_o)
{
  int i = blockIdx.x;
  const float* x = xemb + (size_t)i * (NHEADS * Dn);
  float s0 = 0.0f, s1 = 0.0f;
  for (int c = threadIdx.x; c < NHEADS * Dn; c += 32) {
    float v = x[c];
    s0 += v * Wm[c];
    s1 += v * Wm[NHEADS * Dn + c];
  }
  for (int o = 16; o > 0; o >>= 1) {
    s0 += __shfl_xor(s0, o, 32);
    s1 += __shfl_xor(s1, o, 32);
  }
  if (threadIdx.x == 0) {
    mean_o[i] = s0 + bm[0];
    float lv = fminf(fmaxf(s1 + bm[1], -5.0f), 2.0f);
    std_o[i] = __expf(0.5f * lv);
  }
}

// ---------------------------------------------------------------------------
extern "C" void kernel_launch(void* const* d_in, const int* in_sizes, int n_in,
                              void* d_out, int out_size, void* d_ws, size_t ws_size,
                              hipStream_t stream)
{
  (void)in_sizes; (void)n_in; (void)out_size; (void)ws_size;

  const float* X        = (const float*)d_in[0];
  const float* timec    = (const float*)d_in[1];
  const float* eps1     = (const float*)d_in[2];
  const float* eps2     = (const float*)d_in[3];
  const float* enc_Wih  = (const float*)d_in[4];
  const float* enc_Whh  = (const float*)d_in[5];
  const float* enc_bih  = (const float*)d_in[6];
  const float* enc_bhh  = (const float*)d_in[7];
  const float* mu_W     = (const float*)d_in[8];
  const float* mu_b     = (const float*)d_in[9];
  const float* lv_W     = (const float*)d_in[10];
  const float* lv_b     = (const float*)d_in[11];
  const float* es_now   = (const float*)d_in[12];
  const float* es_lag   = (const float*)d_in[13];
  const float* prior    = (const float*)d_in[14];
  const float* flow_sW  = (const float*)d_in[15];
  const float* flow_sb  = (const float*)d_in[16];
  const float* flow_tW  = (const float*)d_in[17];
  const float* flow_tb  = (const float*)d_in[18];
  const float* tg_Wih   = (const float*)d_in[19];
  /* d_in[20] = tg_Whh : unused by the reference (h0 == 0) */
  const float* tg_bih   = (const float*)d_in[21];
  const float* tg_bhh   = (const float*)d_in[22];
  const float* ltn_W    = (const float*)d_in[23];
  const float* ltn_b    = (const float*)d_in[24];
  const float* gat_Wl   = (const float*)d_in[25];
  const float* gat_bl   = (const float*)d_in[26];
  const float* gat_Wr   = (const float*)d_in[27];
  const float* gat_br   = (const float*)d_in[28];
  const float* gat_att  = (const float*)d_in[29];
  const float* gat_bias = (const float*)d_in[30];
  const float* gl_W     = (const float*)d_in[31];
  const float* gl_b     = (const float*)d_in[32];
  const float* dec_Wih  = (const float*)d_in[33];
  const float* dec_Whh  = (const float*)d_in[34];
  const float* dec_bih  = (const float*)d_in[35];
  const float* dec_bhh  = (const float*)d_in[36];
  const float* fc_W     = (const float*)d_in[37];
  const float* fc_b     = (const float*)d_in[38];

  float* ws = (float*)d_ws;
  // Arena with phase-based aliasing (offsets in floats). Phase 1 (encoder)
  // owns [0, 3145728) as GI_enc; later phases recycle that region.
  float* GI_enc  = ws + 0;         // [16384*192]
  float* e_buf   = ws + 0;         // [4*512*512] GAT scores/alpha (after encoder)
  float* dec_out = ws + 1048576;   // [16384*64]
  float* GI_dec  = ws + 2097152;   // [512*192]
  float* xemb    = ws + 2195456;   // [512*256]
  float* sraw    = ws + 2326528;   // [512*64]
  float* traw    = ws + 2359296;   // [512*64]
  float* gitg    = ws + 2392064;   // [512*192]
  float* hbuf    = ws + 3145728;   // [512*64]  enc_last
  float* Zbuf    = ws + 3178496;   // [512*128]
  float* Z2buf   = ws + 3244032;   // [512*128]
  float* Zf      = ws + 3309568;   // [512*128]
  float* Zn      = ws + 3375104;   // [512*512]
  float* xl      = ws + 3637248;   // [512*256]
  float* xr      = ws + 3768320;   // [512*256]  (arena end: 3899392 floats ~15.6 MB)

  float* out    = (float*)d_out;
  float* recon  = out;                 // [512*32*512]
  float* mu_o   = out + 8388608;       // [512*128]
  float* lv_o   = out + 8454144;       // [512*128]
  float* adjn   = out + 8519680;       // [512*512]
  float* adjl   = out + 8781824;       // [512*512]
  float* mean_o = out + 9043968;       // [512]
  float* std_o  = out + 9044480;       // [512]

  const dim3 w32(32);
  const int EB = 256;

  // ---- Encoder ----
  // GI = X @ enc_Wih.T + bih : [16384,192], K=512 (dominant 3.2 GFLOP GEMM)
  gemm_bias_wmma<<<dim3(G3H / 32, (Bn * Tn) / 32), w32, 0, stream>>>(
      X, Nn, enc_Wih, enc_bih, GI_enc, Bn * Tn, G3H, Nn);
  // Full 32-step recurrence in one launch; final h -> hbuf (= enc_last).
  gru_seq_wmma<<<Bn / 16, 128, 0, stream>>>(
      GI_enc, Tn * G3H, G3H, enc_Whh, enc_bhh, nullptr, hbuf);

  // ---- mu / logvar / reparameterization ----
  gemm_bias_wmma<<<dim3(Ln / 32, Bn / 32), w32, 0, stream>>>(hbuf, Hn, mu_W, mu_b, mu_o, Bn, Ln, Hn);
  gemm_bias_wmma<<<dim3(Ln / 32, Bn / 32), w32, 0, stream>>>(hbuf, Hn, lv_W, lv_b, lv_o, Bn, Ln, Hn);
  reparam<<<(Bn * Ln + EB - 1) / EB, EB, 0, stream>>>(mu_o, lv_o, eps1, eps2, Zbuf, Z2buf, Bn * Ln);

  // ---- Adjacency outputs ----
  adj_kernel<<<(Nn * Nn + EB - 1) / EB, EB, 0, stream>>>(es_now, prior, adjn);
  adj_kernel<<<(Nn * Nn + EB - 1) / EB, EB, 0, stream>>>(es_lag, nullptr, adjl);

  // ---- Flow (z1 = Z[:, :64] via lda=128) + time gate ----
  gemm_bias_wmma<<<dim3(L2n / 32, Bn / 32), w32, 0, stream>>>(Zbuf, Ln, flow_sW, flow_sb, sraw, Bn, L2n, L2n);
  gemm_bias_wmma<<<dim3(L2n / 32, Bn / 32), w32, 0, stream>>>(Zbuf, Ln, flow_tW, flow_tb, traw, Bn, L2n, L2n);
  gemm_bias_wmma<<<dim3(G3H / 32, Bn / 32), w32, 0, stream>>>(timec, Tn, tg_Wih, tg_bih, gitg, Bn, G3H, Tn);
  flow_tg_kernel<<<(Bn * L2n + EB - 1) / EB, EB, 0, stream>>>(Zbuf, sraw, traw, gitg, tg_bhh, Zf);

  // ---- Latent-to-node + GAT ----
  gemm_bias_wmma<<<dim3(Nn / 32, Bn / 32), w32, 0, stream>>>(Zf, Ln, ltn_W, ltn_b, Zn, Bn, Nn, Ln);
  gemm_bias_wmma<<<dim3((NHEADS * Dn) / 32, Bn / 32), w32, 0, stream>>>(Zn, Nn, gat_Wl, gat_bl, xl, Bn, NHEADS * Dn, Nn);
  gemm_bias_wmma<<<dim3((NHEADS * Dn) / 32, Bn / 32), w32, 0, stream>>>(Zn, Nn, gat_Wr, gat_br, xr, Bn, NHEADS * Dn, Nn);
  gat_e_kernel<<<(Bn * Bn * NHEADS + EB - 1) / EB, EB, 0, stream>>>(xl, xr, gat_att, e_buf);
  gat_softmax<<<NHEADS * Bn, w32, 0, stream>>>(e_buf);
  gat_aggregate<<<(Bn * NHEADS * Dn + EB - 1) / EB, EB, 0, stream>>>(e_buf, xl, gat_bias, xemb);
  gl_kernel<<<Bn, w32, 0, stream>>>(xemb, gl_W, gl_b, mean_o, std_o);

  // ---- Decoder (input Z2 identical for all t -> GI computed once) ----
  gemm_bias_wmma<<<dim3(G3H / 32, Bn / 32), w32, 0, stream>>>(Z2buf, Ln, dec_Wih, dec_bih, GI_dec, Bn, G3H, Ln);
  gru_seq_wmma<<<Bn / 16, 128, 0, stream>>>(
      GI_dec, G3H, 0, dec_Whh, dec_bhh, dec_out, nullptr);

  // ---- Output projection: recon_X = dec @ fc_W.T + fc_b (rows are [b][t]) ----
  gemm_bias_wmma<<<dim3(Nn / 32, (Bn * Tn) / 32), w32, 0, stream>>>(
      dec_out, Hn, fc_W, fc_b, recon, Bn * Tn, Nn, Hn);
}
